// STPNR_80573586473192
// MI455X (gfx1250) — matrix operand order repaired and checked by reading, not compiled, not run on hardware
//
#include <hip/hip_runtime.h>
#include <math.h>

typedef __attribute__((ext_vector_type(16))) _Float16 v16h;
typedef __attribute__((ext_vector_type(8)))  float    v8f;

constexpr int NB = 32;      // batch
constexpr int NT = 128;     // time steps
constexpr int NI = 256;     // input dim
constexpr int NH = 256;     // hidden dim
constexpr int NO = 128;     // output dim
constexpr int NF = NH + NI; // 512 fast-weight columns

// ---------------------------------------------------------------------------
// Kernel 1: fast-weight recurrence. One block per batch (batches are fully
// independent through the scan). 1024 threads = 32 waves fill a WGP; each
// wave owns 8 rows of F per timestep. F lives directly in the d_out F_T
// region (16 MB total, L2-resident); h lives in LDS. Per step each F row is
// read ONCE into registers (dot and ||.||^2 computed together), wave32
// shuffle-reduced, then rescaled/updated/written back — one read + one write
// of F per step, which is the bandwidth-roofline minimum. The total-input
// vector chunk for each lane is hoisted into registers once per timestep
// (invariant across the 8 rows), so the inner loop touches only the
// streaming global data.
// ---------------------------------------------------------------------------
__global__ __launch_bounds__(1024)
void stpnr_recurrent(const float* __restrict__ sentence,   // (B,T,I)
                     const float* __restrict__ w_lambda,   // (H,NF)
                     const float* __restrict__ w_gamma,    // (H,NF)
                     const float* __restrict__ weight,     // (H,NF)
                     const float* __restrict__ bias,       // (H)
                     float* __restrict__ hT_out,           // (B,H)
                     float* __restrict__ F_out)            // (B,H,NF)
{
    __shared__ __align__(16) float in_lds[NF];   // [0..NI) = x_t, [NI..NF) = h_{t-1}
    __shared__ float hnew_lds[NH];
    __shared__ float bias_lds[NH];

    const int b    = blockIdx.x;
    const int tid  = threadIdx.x;
    const int lane = tid & 31;
    const int wave = tid >> 5;                   // 0..31

    float* __restrict__ Fb = F_out + (size_t)b * NH * NF;

    // Zero-init this batch's F slice (32768 float4s / 1024 threads).
    {
        float4* F4 = (float4*)Fb;
        const float4 z = make_float4(0.f, 0.f, 0.f, 0.f);
        for (int i = tid; i < NH * NF / 4; i += 1024) F4[i] = z;
    }
    if (tid < NH) {
        in_lds[NI + tid] = 0.f;                  // h_0 = 0
        bias_lds[tid]    = bias[tid];            // bias cached in LDS
    }
    __syncthreads();

    for (int t = 0; t < NT; ++t) {
        // Fresh x_t into LDS.
        if (tid < NI) in_lds[tid] = sentence[((size_t)b * NT + t) * NI + tid];
        __syncthreads();

        // This lane's slice of total_input — invariant across the 8 rows.
        float4 iv[4];
        #pragma unroll
        for (int c = 0; c < 4; ++c)
            iv[c] = *(const float4*)(in_lds + c * 128 + lane * 4);

        #pragma unroll 1
        for (int r = 0; r < 8; ++r) {
            const int row = wave * 8 + r;
            float* __restrict__ Fr = Fb + (size_t)row * NF;
            const float* __restrict__ Wr = weight   + (size_t)row * NF;
            const float* __restrict__ Lr = w_lambda + (size_t)row * NF;
            const float* __restrict__ Gr = w_gamma  + (size_t)row * NF;

            float4 ft[4];
            float dot = 0.f, sq = 0.f;
            #pragma unroll
            for (int c = 0; c < 4; ++c) {
                const int f = c * 128 + lane * 4;
                const float4 fv = *(const float4*)(Fr + f);
                const float4 wv = *(const float4*)(Wr + f);
                ft[c] = fv;
                const float tx = fv.x + wv.x, ty = fv.y + wv.y;
                const float tz = fv.z + wv.z, tw = fv.w + wv.w;
                dot += tx * iv[c].x + ty * iv[c].y + tz * iv[c].z + tw * iv[c].w;
                sq  += tx * tx + ty * ty + tz * tz + tw * tw;
            }
            // wave32 full butterfly reduction (all lanes get the sums).
            #pragma unroll
            for (int m = 16; m >= 1; m >>= 1) {
                dot += __shfl_xor(dot, m, 32);
                sq  += __shfl_xor(sq,  m, 32);
            }
            const float inv_norm = 1.0f / (sqrtf(sq) + 1e-16f);
            const float hnew = tanhf(dot + bias_lds[row]) * inv_norm;
            if (lane == 0) hnew_lds[row] = hnew;

            if (r < 7) __builtin_prefetch(Fr + NF + lane * 4, 0, 0); // next row

            // F_new = lambda * (F/norm) + gamma * (in ⊗ hnew), written in place.
            #pragma unroll
            for (int c = 0; c < 4; ++c) {
                const int f = c * 128 + lane * 4;
                const float4 lv = *(const float4*)(Lr + f);
                const float4 gv = *(const float4*)(Gr + f);
                float4 fn;
                fn.x = lv.x * (ft[c].x * inv_norm) + gv.x * (iv[c].x * hnew);
                fn.y = lv.y * (ft[c].y * inv_norm) + gv.y * (iv[c].y * hnew);
                fn.z = lv.z * (ft[c].z * inv_norm) + gv.z * (iv[c].z * hnew);
                fn.w = lv.w * (ft[c].w * inv_norm) + gv.w * (iv[c].w * hnew);
                *(float4*)(Fr + f) = fn;
            }
        }
        __syncthreads();
        if (tid < NH) in_lds[NI + tid] = hnew_lds[tid];   // h_t feeds next step
        __syncthreads();
    }

    if (tid < NH) hT_out[b * NH + tid] = in_lds[NI + tid];
}

// ---------------------------------------------------------------------------
// Kernel 2: tag_space = h_T @ W_out^T + b_out  via v_wmma_f32_16x16x32_f16.
// 16 output tiles (2 x 8 of 16x16); one wave per tile; K=256 fully unrolled
// into 8 back-to-back WMMA issues so the fragment loads hoist/overlap.
// Fragment layouts per CDNA5 ISA 7.12.2 (wave32):
//   A (16x32 f16):  M = lane%16; half h -> K = h + (h<8?0:8) + (lane<16?0:8)
//   B (32x16 f16):  N = lane%16; half h -> K = h + (h<8?0:8) + (lane<16?0:8)
//   C/D (16x16 f32): VGPR i -> M = i + (lane<16?0:8), N = lane%16
// ---------------------------------------------------------------------------
__global__ __launch_bounds__(512)
void stpnr_out_gemm(const float* __restrict__ hT,     // (B,H) f32
                    const float* __restrict__ W_out,  // (O,H) f32
                    const float* __restrict__ b_out,  // (O)
                    float* __restrict__ tag)          // (B,O)
{
    const int lane  = threadIdx.x & 31;
    const int wv    = threadIdx.x >> 5;   // 0..15
    const int tm    = wv >> 3;            // 0..1  (M tile over batch)
    const int tn    = wv & 7;             // 0..7  (N tile over output)
    const int hi    = (lane >= 16) ? 1 : 0;
    const int lmod  = lane & 15;

    v8f acc = {};
    #pragma unroll
    for (int kc = 0; kc < NH; kc += 32) {
        v16h a, bb;
        #pragma unroll
        for (int h = 0; h < 16; ++h) {
            const int ka = kc + h + (h < 8 ? 0 : 8) + (hi ? 8 : 0);
            a[h] = (_Float16)hT[(tm * 16 + lmod) * NH + ka];
            const int kb = kc + h + (h < 8 ? 0 : 8) + (hi ? 8 : 0);
            bb[h] = (_Float16)W_out[(tn * 16 + lmod) * NH + kb];
        }
        acc = __builtin_amdgcn_wmma_f32_16x16x32_f16(
            /*neg_a=*/false, a, /*neg_b=*/false, bb,
            /*c_mod=*/(short)0, acc, /*reuse_a=*/false, /*reuse_b=*/false);
    }

    #pragma unroll
    for (int i = 0; i < 8; ++i) {
        const int m = tm * 16 + i + (hi ? 8 : 0);
        const int n = tn * 16 + lmod;
        tag[m * NO + n] = acc[i] + b_out[n];
    }
}

// ---------------------------------------------------------------------------
// d_out layout (reference return order): tag_space (32*128) | h_T (32*256) |
// F_T (32*256*512). F_T doubles as the recurrence state, zero-initialized
// in-kernel each call -> deterministic under graph replay.
// ---------------------------------------------------------------------------
extern "C" void kernel_launch(void* const* d_in, const int* in_sizes, int n_in,
                              void* d_out, int out_size, void* d_ws, size_t ws_size,
                              hipStream_t stream) {
    const float* sentence = (const float*)d_in[0];
    const float* w_lambda = (const float*)d_in[1];
    const float* w_gamma  = (const float*)d_in[2];
    const float* weight   = (const float*)d_in[3];
    const float* bias     = (const float*)d_in[4];
    const float* W_out    = (const float*)d_in[5];
    const float* b_out    = (const float*)d_in[6];

    float* out = (float*)d_out;
    float* tag = out;                       // (B,O)
    float* hT  = out + NB * NO;             // (B,H)
    float* Fo  = out + NB * NO + NB * NH;   // (B,H,NF)

    stpnr_recurrent<<<NB, 1024, 0, stream>>>(sentence, w_lambda, w_gamma,
                                             weight, bias, hT, Fo);
    stpnr_out_gemm<<<1, 512, 0, stream>>>(hT, W_out, b_out, tag);
}
